// QuickshiftGroups_14697378087159
// MI455X (gfx1250) — compile-verified, matrix operand order checked by reading, not run on hardware
//
#include <hip/hip_runtime.h>
#include <cmath>

typedef __attribute__((ext_vector_type(16))) _Float16 v16h;
typedef __attribute__((ext_vector_type(8)))  float    v8f;

#define BATCH 8
#define HH 512
#define WW 512
#define NPIX (HH*WW)        // 262144 = 1<<18
#define RWIN 10

union alignas(8) H4 { struct { unsigned lo, hi; } u; _Float16 h[4]; };

__device__ __forceinline__ int symref(int m, int n) {
  if (m < 0) m = -1 - m;
  if (m >= n) m = 2 * n - 1 - m;
  return m;
}

__device__ __forceinline__ unsigned long long shflxor_u64(unsigned long long v, int m) {
  unsigned lo = (unsigned)v, hi = (unsigned)(v >> 32);
  lo = __shfl_xor(lo, m, 32);
  hi = __shfl_xor(hi, m, 32);
  return ((unsigned long long)hi << 32) | lo;
}

// ---------------- Gaussian blur (sigma=1, r=3, symmetric pad) ----------------
__global__ __launch_bounds__(256)
void blur_v_kernel(const float* __restrict__ in, float* __restrict__ out,
                   float w0, float w1, float w2, float w3) {
  int tid = blockIdx.x * 256 + threadIdx.x;      // over BATCH*3*NPIX
  int x  = tid & (WW - 1);
  int y  = (tid >> 9) & (HH - 1);
  int pl = tid >> 18;                            // plane 0..23
  const float* p = in + ((size_t)pl << 18);
  float v = w0 * p[(y << 9) | x];
  v += w1 * (p[(symref(y - 1, HH) << 9) | x] + p[(symref(y + 1, HH) << 9) | x]);
  v += w2 * (p[(symref(y - 2, HH) << 9) | x] + p[(symref(y + 2, HH) << 9) | x]);
  v += w3 * (p[(symref(y - 3, HH) << 9) | x] + p[(symref(y + 3, HH) << 9) | x]);
  out[tid] = v;
}

__global__ __launch_bounds__(256)
void blur_h_pack_kernel(const float* __restrict__ tmp, _Float16* __restrict__ feat,
                        float w0, float w1, float w2, float w3) {
  int tid = blockIdx.x * 256 + threadIdx.x;      // over BATCH*NPIX
  int x   = tid & (WW - 1);
  int y   = (tid >> 9) & (HH - 1);
  int img = tid >> 18;
  H4 o;
#pragma unroll
  for (int ch = 0; ch < 3; ++ch) {
    const float* p = tmp + (((size_t)(img * 3 + ch)) << 18) + ((size_t)y << 9);
    float v = w0 * p[x];
    v += w1 * (p[symref(x - 1, WW)] + p[symref(x + 1, WW)]);
    v += w2 * (p[symref(x - 2, WW)] + p[symref(x + 2, WW)]);
    v += w3 * (p[symref(x - 3, WW)] + p[symref(x + 3, WW)]);
    o.h[ch] = (_Float16)v;
  }
  o.h[3] = (_Float16)0.0f;
  *(H4*)(feat + (size_t)tid * 4) = o;
}

// ---------------- Density pass: WMMA for cross terms, exp accumulation -------
// one wave32 = one 16-pixel row tile. 21 dy x 3 WMMA tiles cover all offsets.
__global__ __launch_bounds__(256)
void density_kernel(const _Float16* __restrict__ feat, float* __restrict__ density) {
  const int lane = threadIdx.x;
  const int tile = blockIdx.x * 8 + threadIdx.y;
  const int img  = tile >> 14;                   // 512*32 tiles per image
  const int rem  = tile & 16383;
  const int y    = rem >> 5;
  const int x0   = (rem & 31) << 4;
  const _Float16* fimg = feat + (((size_t)img) << 18) * 4;

  // A operand: lanes 0-15 hold pixel (y, x0+lane) channels in K=0..2
  v16h a;
#pragma unroll
  for (int t = 0; t < 16; ++t) a[t] = (_Float16)0.0f;
  float nlane = 0.0f;
  if (lane < 16) {
    H4 p = *(const H4*)(fimg + (((size_t)y << 9) + x0 + lane) * 4);
    a[0] = p.h[0]; a[1] = p.h[1]; a[2] = p.h[2];
    float c0 = (float)p.h[0], c1 = (float)p.h[1], c2 = (float)p.h[2];
    nlane = c0 * c0 + c1 * c1 + c2 * c2;
  }
  float ni[8];
#pragma unroll
  for (int r = 0; r < 8; ++r) ni[r] = __shfl(nlane, (lane < 16) ? r : 8 + r, 32);

  float acc[8];
#pragma unroll
  for (int r = 0; r < 8; ++r) acc[r] = 0.0f;

  const float inv2k2 = 1.0f / 18.0f;
  const int colbase = x0 - RWIN + (lane & 15);

  for (int dy = -RWIN; dy <= RWIN; ++dy) {
    int yy = y + dy;
    bool rowok = (yy >= 0) && (yy < HH);
    int yyc = rowok ? yy : 0;
    float spdy = (float)(dy * dy);
#pragma unroll
    for (int k = 0; k < 3; ++k) {
      int col = colbase + (k << 4);
      bool colok = (col >= 0) && (col < WW);
      int colc = colok ? col : 0;
      H4 q = *(const H4*)(fimg + (((size_t)yyc << 9) + colc) * 4);
      bool ok = rowok && colok;
      float q0 = ok ? (float)q.h[0] : 0.0f;
      float q1 = ok ? (float)q.h[1] : 0.0f;
      float q2 = ok ? (float)q.h[2] : 0.0f;
      float ncol = q0 * q0 + q1 * q1 + q2 * q2;
      v16h b;
#pragma unroll
      for (int t = 0; t < 16; ++t) b[t] = (_Float16)0.0f;
      if (lane < 16 && ok) { b[0] = q.h[0]; b[1] = q.h[1]; b[2] = q.h[2]; }
      v8f c;
#pragma unroll
      for (int t = 0; t < 8; ++t) c[t] = 0.0f;
      c = __builtin_amdgcn_wmma_f32_16x16x32_f16(false, a, false, b, (short)0, c, false, false);
#pragma unroll
      for (int r = 0; r < 8; ++r) {
        int px = x0 + ((lane < 16) ? r : 8 + r);
        int dx = col - px;
        bool v = ok && (dx >= -RWIN) && (dx <= RWIN);
        float d2 = ni[r] + ncol - 2.0f * c[r] + spdy + (float)(dx * dx);
        d2 = fmaxf(d2, 0.0f);
        acc[r] += __expf(v ? (-d2 * inv2k2) : -1.0e30f);
      }
    }
  }
#pragma unroll
  for (int r = 0; r < 8; ++r) {
    float v = acc[r];
    v += __shfl_xor(v, 1, 32);
    v += __shfl_xor(v, 2, 32);
    v += __shfl_xor(v, 4, 32);
    v += __shfl_xor(v, 8, 32);
    acc[r] = v;
  }
  float* dimg = density + ((size_t)img << 18);
#pragma unroll
  for (int r = 0; r < 8; ++r) {
    int px = x0 + ((lane < 16) ? r : 8 + r);
    if ((lane & 15) == r) dimg[((size_t)y << 9) + px] = acc[r];
  }
}

// ---------------- Parent pass: min (dist2, scan-order) over higher-density nb
__global__ __launch_bounds__(256)
void parent_kernel(const _Float16* __restrict__ feat, const float* __restrict__ density,
                   int* __restrict__ parent) {
  const int lane = threadIdx.x;
  const int tile = blockIdx.x * 8 + threadIdx.y;
  const int img  = tile >> 14;
  const int rem  = tile & 16383;
  const int y    = rem >> 5;
  const int x0   = (rem & 31) << 4;
  const _Float16* fimg = feat + (((size_t)img) << 18) * 4;
  const float* dimg = density + ((size_t)img << 18);

  v16h a;
#pragma unroll
  for (int t = 0; t < 16; ++t) a[t] = (_Float16)0.0f;
  float nlane = 0.0f, dlane = 0.0f;
  if (lane < 16) {
    H4 p = *(const H4*)(fimg + (((size_t)y << 9) + x0 + lane) * 4);
    a[0] = p.h[0]; a[1] = p.h[1]; a[2] = p.h[2];
    float c0 = (float)p.h[0], c1 = (float)p.h[1], c2 = (float)p.h[2];
    nlane = c0 * c0 + c1 * c1 + c2 * c2;
    dlane = dimg[((size_t)y << 9) + x0 + lane];
  }
  float ni[8], di[8];
#pragma unroll
  for (int r = 0; r < 8; ++r) {
    int src = (lane < 16) ? r : 8 + r;
    ni[r] = __shfl(nlane, src, 32);
    di[r] = __shfl(dlane, src, 32);
  }

  unsigned long long key[8];
#pragma unroll
  for (int r = 0; r < 8; ++r)
    key[r] = (((unsigned long long)0x7F800000u) << 32) | 1023u;   // (+inf, big ord)

  const int colbase = x0 - RWIN + (lane & 15);

  for (int dy = -RWIN; dy <= RWIN; ++dy) {
    int yy = y + dy;
    bool rowok = (yy >= 0) && (yy < HH);
    int yyc = rowok ? yy : 0;
    float spdy = (float)(dy * dy);
    int ordBase = (dy + RWIN) * 21;
#pragma unroll
    for (int k = 0; k < 3; ++k) {
      int col = colbase + (k << 4);
      bool colok = (col >= 0) && (col < WW);
      int colc = colok ? col : 0;
      H4 q = *(const H4*)(fimg + (((size_t)yyc << 9) + colc) * 4);
      bool ok = rowok && colok;
      float q0 = ok ? (float)q.h[0] : 0.0f;
      float q1 = ok ? (float)q.h[1] : 0.0f;
      float q2 = ok ? (float)q.h[2] : 0.0f;
      float ncol = q0 * q0 + q1 * q1 + q2 * q2;
      float dn = ok ? dimg[((size_t)yyc << 9) + colc] : -1.0e30f;
      v16h b;
#pragma unroll
      for (int t = 0; t < 16; ++t) b[t] = (_Float16)0.0f;
      if (lane < 16 && ok) { b[0] = q.h[0]; b[1] = q.h[1]; b[2] = q.h[2]; }
      v8f c;
#pragma unroll
      for (int t = 0; t < 8; ++t) c[t] = 0.0f;
      c = __builtin_amdgcn_wmma_f32_16x16x32_f16(false, a, false, b, (short)0, c, false, false);
#pragma unroll
      for (int r = 0; r < 8; ++r) {
        int px = x0 + ((lane < 16) ? r : 8 + r);
        int dx = col - px;
        bool v = ok && (dx >= -RWIN) && (dx <= RWIN);
        float d2 = ni[r] + ncol - 2.0f * c[r] + spdy + (float)(dx * dx);
        d2 = fmaxf(d2, 0.0f);
        bool take = v && (dn > di[r]);
        unsigned long long cand = take
            ? ((((unsigned long long)__float_as_uint(d2)) << 32) | (unsigned)(ordBase + dx + RWIN))
            : 0xFFFFFFFFFFFFFFFFull;
        key[r] = (cand < key[r]) ? cand : key[r];
      }
    }
  }
#pragma unroll
  for (int r = 0; r < 8; ++r) {
    unsigned long long v = key[r];
    unsigned long long w;
    w = shflxor_u64(v, 1); v = (w < v) ? w : v;
    w = shflxor_u64(v, 2); v = (w < v) ? w : v;
    w = shflxor_u64(v, 4); v = (w < v) ? w : v;
    w = shflxor_u64(v, 8); v = (w < v) ? w : v;
    key[r] = v;
  }
  int* pimg = parent + ((size_t)img << 18);
#pragma unroll
  for (int r = 0; r < 8; ++r) {
    int px = x0 + ((lane < 16) ? r : 8 + r);
    if ((lane & 15) == r) {
      unsigned long long kk = key[r];
      float bd = __uint_as_float((unsigned)(kk >> 32));
      int self = (y << 9) + px;
      int par;
      if (sqrtf(bd) > 10.0f) {
        par = self;
      } else {
        int ord = (int)(kk & 0xFFFFFFFFull);
        int qy = ord / 21;
        int ddy = qy - RWIN;
        int ddx = (ord - qy * 21) - RWIN;
        par = ((y + ddy) << 9) + (px + ddx);
      }
      pimg[self] = par;
    }
  }
}

// ---------------- Root chase (density strictly increases -> acyclic) --------
__global__ __launch_bounds__(256)
void root_kernel(const int* __restrict__ parent, int* __restrict__ root) {
  int tid = blockIdx.x * 256 + threadIdx.x;
  int img = tid >> 18;
  int i   = tid & (NPIX - 1);
  const int* p = parent + ((size_t)img << 18);
  int cur = i;
  for (int it = 0; it < NPIX; ++it) {
    int nx = p[cur];
    if (nx == cur) break;
    cur = nx;
  }
  root[tid] = cur;
}

// ---------------- Compact relabel: exclusive prefix count of roots ----------
__global__ __launch_bounds__(256)
void blocksum_kernel(const int* __restrict__ root, int* __restrict__ blocksum) {
  int blk = blockIdx.x, t = threadIdx.x;
  int img = blk >> 8, cb = blk & 255;
  const int* rimg = root + ((size_t)img << 18);
  int base = (cb << 10) + (t << 2);
  int s0 = (rimg[base] == base) + (rimg[base + 1] == base + 1) +
           (rimg[base + 2] == base + 2) + (rimg[base + 3] == base + 3);
  __shared__ int sh[256];
  sh[t] = s0; __syncthreads();
  for (int off = 128; off > 0; off >>= 1) {
    if (t < off) sh[t] += sh[t + off];
    __syncthreads();
  }
  if (t == 0) blocksum[blk] = sh[0];
}

__global__ __launch_bounds__(256)
void scan_blocks_kernel(const int* __restrict__ blocksum, int* __restrict__ blockoff,
                        int* __restrict__ nseg) {
  __shared__ int sh[256];
  int img = blockIdx.x, t = threadIdx.x;
  int v = blocksum[img * 256 + t];
  sh[t] = v; __syncthreads();
  for (int off = 1; off < 256; off <<= 1) {
    int add = (t >= off) ? sh[t - off] : 0;
    __syncthreads();
    sh[t] += add;
    __syncthreads();
  }
  blockoff[img * 256 + t] = sh[t] - v;
  if (t == 255) nseg[img] = sh[255];
}

__global__ __launch_bounds__(256)
void rank_kernel(const int* __restrict__ root, const int* __restrict__ blockoff,
                 int* __restrict__ rank) {
  int blk = blockIdx.x, t = threadIdx.x;
  int img = blk >> 8, cb = blk & 255;
  const int* rimg = root + ((size_t)img << 18);
  int base = (cb << 10) + (t << 2);
  int c0 = (rimg[base] == base), c1 = (rimg[base + 1] == base + 1);
  int c2 = (rimg[base + 2] == base + 2), c3 = (rimg[base + 3] == base + 3);
  int tsum = c0 + c1 + c2 + c3;
  __shared__ int sh[256];
  sh[t] = tsum; __syncthreads();
  for (int off = 1; off < 256; off <<= 1) {
    int add = (t >= off) ? sh[t - off] : 0;
    __syncthreads();
    sh[t] += add;
    __syncthreads();
  }
  int off0 = blockoff[blk] + (sh[t] - tsum);
  int* rk = rank + ((size_t)img << 18) + base;
  rk[0] = off0; off0 += c0;
  rk[1] = off0; off0 += c1;
  rk[2] = off0; off0 += c2;
  rk[3] = off0;
}

// ---------------- Cap to 16 groups + one-hot [B,16,H,W] ---------------------
__global__ __launch_bounds__(256)
void onehot_kernel(const int* __restrict__ root, const int* __restrict__ rank,
                   const int* __restrict__ nseg, float* __restrict__ out) {
  int tid = blockIdx.x * 256 + threadIdx.x;
  int img = tid >> 18;
  int i   = tid & (NPIX - 1);
  int r   = root[tid];
  int lbl = rank[((size_t)img << 18) + r];
  int ns  = nseg[img];
  if (ns > 16) {
    float div = (float)ns * (1.0f / 16.0f);
    lbl = (int)floorf((float)lbl / div);
  }
  lbl = min(max(lbl, 0), 15);
  float* o = out + (((size_t)img) << 22) + i;
#pragma unroll
  for (int c = 0; c < 16; ++c) o[(size_t)c << 18] = (c == lbl) ? 1.0f : 0.0f;
}

extern "C" void kernel_launch(void* const* d_in, const int* in_sizes, int n_in,
                              void* d_out, int out_size, void* d_ws, size_t ws_size,
                              hipStream_t stream) {
  (void)in_sizes; (void)n_in; (void)out_size;
  const float* x = (const float*)d_in[0];
  float* out = (float*)d_out;

  // Gaussian weights, sigma=1, r=3, normalized
  float g1 = expf(-0.5f), g2 = expf(-2.0f), g3 = expf(-4.5f);
  float s = 1.0f + 2.0f * (g1 + g2 + g3);
  float w0 = 1.0f / s, w1 = g1 / s, w2 = g2 / s, w3 = g3 / s;

  char* wsb = (char*)d_ws;
  size_t planeBytes = (size_t)BATCH * NPIX * 4;       // 8 MB per int/float plane
  size_t featBytes  = 2 * planeBytes;                  // packed half4 features
  size_t needed = featBytes + 3 * planeBytes;          // ~42 MB
  if (ws_size < needed) return;

  _Float16* feat   = (_Float16*)wsb;
  float*    tmp    = (float*)(wsb + featBytes);        // blur temp (3 planes)
  float*    densit = (float*)(wsb + featBytes);        // reuses tmp after blur
  int*      parent = (int*)(wsb + featBytes + planeBytes);
  int*      root   = (int*)(wsb + featBytes + 2 * planeBytes);
  int*      rank   = (int*)wsb;                        // reuses feat after parent pass
  int*      blocksum = (int*)(wsb + planeBytes);       // also inside dead feat region
  int*      blockoff = blocksum + BATCH * 256;
  int*      nseg     = blockoff + BATCH * 256;

  blur_v_kernel<<<BATCH * 3 * NPIX / 256, 256, 0, stream>>>(x, tmp, w0, w1, w2, w3);
  blur_h_pack_kernel<<<BATCH * NPIX / 256, 256, 0, stream>>>(tmp, feat, w0, w1, w2, w3);

  dim3 wblk(32, 8);
  int ntiles = BATCH * HH * (WW / 16);                 // 131072 wave-tiles
  density_kernel<<<ntiles / 8, wblk, 0, stream>>>(feat, densit);
  parent_kernel<<<ntiles / 8, wblk, 0, stream>>>(feat, densit, parent);

  root_kernel<<<BATCH * NPIX / 256, 256, 0, stream>>>(parent, root);
  blocksum_kernel<<<BATCH * 256, 256, 0, stream>>>(root, blocksum);
  scan_blocks_kernel<<<BATCH, 256, 0, stream>>>(blocksum, blockoff, nseg);
  rank_kernel<<<BATCH * 256, 256, 0, stream>>>(root, blockoff, rank);
  onehot_kernel<<<BATCH * NPIX / 256, 256, 0, stream>>>(root, rank, nseg, out);
}
